// AxisNetFusion_83253646066077
// MI455X (gfx1250) — matrix operand order repaired and checked by reading, not compiled, not run on hardware
//
#include <hip/hip_runtime.h>
#include <math.h>

typedef _Float16 half_t;
typedef __attribute__((ext_vector_type(16))) _Float16 v16h;
typedef __attribute__((ext_vector_type(8)))  _Float16 v8h;
typedef __attribute__((ext_vector_type(8)))  float    v8f;
typedef __attribute__((ext_vector_type(4)))  float    v4f;

// ---------------- WMMA helpers (CDNA5 v_wmma_f32_16x16x32_f16) ----------------

static __device__ inline v8f wmma16(v16h a, v16h b, v8f c) {
  return __builtin_amdgcn_wmma_f32_16x16x32_f16(
      /*neg_a=*/false, a, /*neg_b=*/false, b,
      /*c_mod=*/(short)0, c, /*reuse_a=*/false, /*reuse_b=*/false);
}

// A fragment (16x32 f16, one K-chunk) from a contiguous, 16B-aligned half row.
// ISA layout: lane holds row M=lane&15; p in [0,8): K = kbase+klo+p ; p in [8,16): K = kbase+klo+16+(p-8),
// klo = (lane>=16 ? 8 : 0).  Both 8-half groups are contiguous -> two b128 loads.
static __device__ inline v16h a_frag(const half_t* rowp, int kbase, bool hi) {
  const half_t* p0 = rowp + kbase + (hi ? 8 : 0);
  union { v16h v; v8h h[2]; } u;
  u.h[0] = *(const v8h*)(p0);
  u.h[1] = *(const v8h*)(p0 + 16);
  return u.v;
}

// B fragment (32x16 f16) from TRANSPOSED weights Wt[N][K] (f16).
// ISA layout: lane holds col N=lane&15; p in [0,8): K = kc*32+klo+p ; p in [8,16): K = kc*32+klo+16+(p-8),
// klo = (lane>=16 ? 8 : 0).  Both 8-half groups are contiguous in the transposed layout -> two b128 loads.
static __device__ inline v16h b_frag_t(const half_t* Wt, int K, int n, int kc, bool hi) {
  const half_t* p0 = Wt + (size_t)n * K + kc * 32 + (hi ? 8 : 0);
  union { v16h v; v8h h[2]; } u;
  u.h[0] = *(const v8h*)(p0);
  u.h[1] = *(const v8h*)(p0 + 16);
  return u.v;
}

static __device__ inline void atomic_add_f32(float* p, float v) {
  unsafeAtomicAdd(p, v);  // global_atomic_add_f32 on gfx1250
}

// ---------------- elementwise utility kernels ----------------

__global__ void k_f32_to_f16(const float* __restrict__ src, half_t* __restrict__ dst, long long n) {
  long long i = (long long)blockIdx.x * blockDim.x + threadIdx.x;
  if (i < n) dst[i] = (half_t)src[i];
}

// dst[mat][n][k] = (f16) src[mat][k][n]   (batched transpose-convert; dst index == i)
__global__ void k_transpose_f16(const float* __restrict__ src, half_t* __restrict__ dst,
                                int K, int NC, long long total) {
  long long i = (long long)blockIdx.x * blockDim.x + threadIdx.x;
  if (i >= total) return;
  long long mk = (long long)K * NC;
  long long mat = i / mk;
  long long rem = i - mat * mk;
  int n = (int)(rem / K);
  int k = (int)(rem - (long long)n * K);
  dst[i] = (half_t)src[mat * mk + (long long)k * NC + n];
}

__global__ void k_deg_scatter(const int* __restrict__ row, const float* __restrict__ ew,
                              float* __restrict__ deg, int E) {
  int e = blockIdx.x * blockDim.x + threadIdx.x;
  if (e < E) atomic_add_f32(&deg[row[e]], ew[e]);
}

__global__ void k_dis(const float* __restrict__ deg, float* __restrict__ dis, int N) {
  int n = blockIdx.x * blockDim.x + threadIdx.x;
  if (n < N) {
    float d = deg[n];
    dis[n] = (d > 0.f) ? rsqrtf(fmaxf(d, 1e-30f)) : 0.f;
  }
}

__global__ void k_norm(const int* __restrict__ row, const int* __restrict__ col,
                       const float* __restrict__ ew, const float* __restrict__ dis,
                       float* __restrict__ nv, int E) {
  int e = blockIdx.x * blockDim.x + threadIdx.x;
  if (e < E) nv[e] = -dis[row[e]] * ew[e] * dis[col[e]];
}

// out[col[e]] += norm[e] * src[row[e]] ; src is half or float, 4 features/thread
template <typename T>
__global__ void k_prop_scatter(const int* __restrict__ row, const int* __restrict__ col,
                               const float* __restrict__ nv, const T* __restrict__ src,
                               int srcStride, float* __restrict__ dst, int E) {
  long long idx = (long long)blockIdx.x * blockDim.x + threadIdx.x;
  int e = (int)(idx >> 6);
  if (e >= E) return;
  int f = (int)(idx & 63) << 2;
  float w = nv[e];
  const T* sp = src + (size_t)row[e] * srcStride + f;
  float* dp = dst + (size_t)col[e] * 256 + f;
#pragma unroll
  for (int i = 0; i < 4; ++i) atomic_add_f32(dp + i, w * (float)sp[i]);
}

// t1_16 = (half)t1_32 ; t2_16 = (half)(2*t2_32 - (float)src16)
__global__ void k_cheb_finalize(const float* __restrict__ t1_32, const float* __restrict__ t2_32,
                                const half_t* __restrict__ src16, int srcStride,
                                half_t* __restrict__ t1_16, half_t* __restrict__ t2_16,
                                long long total) {
  long long i = (long long)blockIdx.x * blockDim.x + threadIdx.x;
  if (i >= total) return;
  long long r = i >> 8;
  int c = (int)(i & 255);
  t1_16[i] = (half_t)t1_32[i];
  float tx0 = (float)src16[(size_t)r * srcStride + c];
  t2_16[i] = (half_t)(2.f * t2_32[i] - tx0);
}

// ---------------- edge encoder: PAE MLP + cosine, WMMA tiles of 16 edges ----------------
// W1t: [128][32] f16 transposed, W2t: [128][128] f16 transposed.
// Both halves processed together so every B fragment is loaded once and consumed immediately
// (nothing loop-invariant left to hoist/spill).

#define EW_WAVES 2
__global__ void k_edge_encoder(const float* __restrict__ enin, const half_t* __restrict__ W1t,
                               const float* __restrict__ b1, const float* __restrict__ g1,
                               const float* __restrict__ be1, const half_t* __restrict__ W2t,
                               const float* __restrict__ b2, float* __restrict__ ew, int E) {
  __shared__ half_t sh_h[EW_WAVES][2][16 * 128];  // BN'd layer-1 activations, both halves (f16)
  __shared__ float  sh_g[EW_WAVES][2][16 * 128];  // layer-2 outputs, both halves (f32)
  int lane = threadIdx.x & 31;
  int wv = threadIdx.x >> 5;
  bool hi = lane >= 16;
  int l15 = lane & 15;
  int tile = blockIdx.x * EW_WAVES + wv;
  int ebase = tile * 16;
  int erow = ebase + l15;
  bool evalid = erow < E;
  const float bnsc = rsqrtf(1.0f + 1e-5f);

  // ---- layer-1 A fragments for both halves (vector fp32 loads -> f16) ----
  v16h a1[2];
#pragma unroll
  for (int hsel = 0; hsel < 2; ++hsel) {
    const float* xr = enin + (size_t)(evalid ? erow : 0) * 64 + hsel * 32 + (hi ? 8 : 0);
    v4f x0 = *(const v4f*)(xr);
    v4f x1 = *(const v4f*)(xr + 4);
    v4f x2 = *(const v4f*)(xr + 16);
    v4f x3 = *(const v4f*)(xr + 20);
#pragma unroll
    for (int p = 0; p < 4; ++p) {
      a1[hsel][p]      = evalid ? (half_t)x0[p] : (half_t)0.f;
      a1[hsel][p + 4]  = evalid ? (half_t)x1[p] : (half_t)0.f;
      a1[hsel][p + 8]  = evalid ? (half_t)x2[p] : (half_t)0.f;
      a1[hsel][p + 12] = evalid ? (half_t)x3[p] : (half_t)0.f;
    }
  }

  // ---- layer 1: h = BN(relu(x@W1+b1)) -> LDS; B fragment shared by both halves ----
#pragma unroll
  for (int cb = 0; cb < 8; ++cb) {
    int colc = cb * 16 + l15;
    v16h b = b_frag_t(W1t, 32, colc, 0, hi);
    float sc = g1[colc] * bnsc, bb = b1[colc], bo = be1[colc];
#pragma unroll
    for (int hsel = 0; hsel < 2; ++hsel) {
      v8f c = {0.f, 0.f, 0.f, 0.f, 0.f, 0.f, 0.f, 0.f};
      c = wmma16(a1[hsel], b, c);
#pragma unroll
      for (int r = 0; r < 8; ++r) {
        int rr = hi ? r + 8 : r;
        float v = fmaxf(c[r] + bb, 0.f) * sc + bo;
        sh_h[wv][hsel][rr * 128 + colc] = (half_t)v;
      }
    }
  }
  __syncthreads();

  // ---- layer 2: g = h @ W2 + b2 ; A fragments from LDS, B shared by both halves ----
  v16h a2[2][4];
#pragma unroll
  for (int hsel = 0; hsel < 2; ++hsel)
#pragma unroll
    for (int kc = 0; kc < 4; ++kc)
      a2[hsel][kc] = a_frag(&sh_h[wv][hsel][l15 * 128], kc * 32, hi);

#pragma unroll
  for (int cb = 0; cb < 8; ++cb) {
    int colc = cb * 16 + l15;
    v8f acc0 = {0.f, 0.f, 0.f, 0.f, 0.f, 0.f, 0.f, 0.f};
    v8f acc1 = {0.f, 0.f, 0.f, 0.f, 0.f, 0.f, 0.f, 0.f};
#pragma unroll
    for (int kc = 0; kc < 4; ++kc) {
      v16h b = b_frag_t(W2t, 128, colc, kc, hi);
      acc0 = wmma16(a2[0][kc], b, acc0);
      acc1 = wmma16(a2[1][kc], b, acc1);
    }
    float bb = b2[colc];
#pragma unroll
    for (int r = 0; r < 8; ++r) {
      int rr = hi ? r + 8 : r;
      sh_g[wv][0][rr * 128 + colc] = acc0[r] + bb;
      sh_g[wv][1][rr * 128 + colc] = acc1[r] + bb;
    }
  }
  __syncthreads();

  // ---- cosine similarity reduction: one edge per lane (lanes 0..15), vector LDS reads ----
  if (lane < 16) {
    const v4f* gg1 = (const v4f*)&sh_g[wv][0][lane * 128];
    const v4f* gg2 = (const v4f*)&sh_g[wv][1][lane * 128];
    float dot = 0.f, s1 = 0.f, s2 = 0.f;
#pragma unroll 8
    for (int j = 0; j < 32; ++j) {
      v4f u = gg1[j], v = gg2[j];
#pragma unroll
      for (int q = 0; q < 4; ++q) {
        dot += u[q] * v[q]; s1 += u[q] * u[q]; s2 += v[q] * v[q];
      }
    }
    float n1 = fmaxf(sqrtf(s1), 1e-8f);
    float n2 = fmaxf(sqrtf(s2), 1e-8f);
    int e = ebase + lane;
    if (e < E) ew[e] = (dot / (n1 * n2) + 1.f) * 0.5f;
  }
}

// ---------------- ChebConv combine: h = relu(Tx0@W0 + Tx1@W1 + Tx2@W2) ----------------
// block = 256 threads (8 independent waves), each wave a 16-node tile; blockIdx.y = 16-col block.
// W3t = three transposed 256x256 f16 matrices, contiguous.

__global__ void k_cheb_gemm(const half_t* __restrict__ src0, int stride0,
                            const half_t* __restrict__ t1, const half_t* __restrict__ t2,
                            const half_t* __restrict__ W3t, half_t* __restrict__ outp,
                            int outStride, int N) {
  int lane = threadIdx.x & 31;
  int wv = threadIdx.x >> 5;
  bool hi = lane >= 16;
  int l15 = lane & 15;
  int mbase = blockIdx.x * 128 + wv * 16;
  int row = mbase + l15;
  size_t r0 = (size_t)(row < N ? row : 0);  // clamp: WMMA rows independent, stores guarded
  const half_t* rp0 = src0 + r0 * stride0;
  const half_t* rp1 = t1 + r0 * 256;
  const half_t* rp2 = t2 + r0 * 256;
  int n = blockIdx.y * 16 + l15;
  v8f acc = {0.f, 0.f, 0.f, 0.f, 0.f, 0.f, 0.f, 0.f};
#pragma unroll
  for (int kc = 0; kc < 8; ++kc)
    acc = wmma16(a_frag(rp0, kc * 32, hi), b_frag_t(W3t, 256, n, kc, hi), acc);
#pragma unroll
  for (int kc = 0; kc < 8; ++kc)
    acc = wmma16(a_frag(rp1, kc * 32, hi), b_frag_t(W3t + 65536, 256, n, kc, hi), acc);
#pragma unroll
  for (int kc = 0; kc < 8; ++kc)
    acc = wmma16(a_frag(rp2, kc * 32, hi), b_frag_t(W3t + 131072, 256, n, kc, hi), acc);
#pragma unroll
  for (int r = 0; r < 8; ++r) {
    int rr = mbase + (hi ? r + 8 : r);
    if (rr < N) outp[(size_t)rr * outStride + n] = (half_t)fmaxf(acc[r], 0.f);
  }
}

// ---------------- cls head GEMM: z = BN(relu(jk @ W1 + b1)) ----------------
// Wt = transposed cls_w1: [256][1024] f16.

__global__ void k_cls_gemm(const half_t* __restrict__ jk, const half_t* __restrict__ Wt,
                           const float* __restrict__ b1, const float* __restrict__ g,
                           const float* __restrict__ bb, half_t* __restrict__ z, int N) {
  int lane = threadIdx.x & 31;
  int wv = threadIdx.x >> 5;
  bool hi = lane >= 16;
  int l15 = lane & 15;
  int mbase = blockIdx.x * 128 + wv * 16;
  int row = mbase + l15;
  size_t r0 = (size_t)(row < N ? row : 0);
  const half_t* rp = jk + r0 * 1024;
  int n = blockIdx.y * 16 + l15;
  v8f acc = {0.f, 0.f, 0.f, 0.f, 0.f, 0.f, 0.f, 0.f};
#pragma unroll
  for (int kc = 0; kc < 32; ++kc)
    acc = wmma16(a_frag(rp, kc * 32, hi), b_frag_t(Wt, 1024, n, kc, hi), acc);
  const float bnsc = rsqrtf(1.0f + 1e-5f);
  float sc = g[n] * bnsc, b1v = b1[n], bv = bb[n];
#pragma unroll
  for (int r = 0; r < 8; ++r) {
    int rr = mbase + (hi ? r + 8 : r);
    if (rr < N) z[(size_t)rr * 256 + n] = (half_t)(fmaxf(acc[r] + b1v, 0.f) * sc + bv);
  }
}

// logit = z @ W2 + b2  (256 x 2, trivial)
__global__ void k_cls_final(const half_t* __restrict__ z, const float* __restrict__ w2,
                            const float* __restrict__ b2, float* __restrict__ out, int N) {
  int n = blockIdx.x * blockDim.x + threadIdx.x;
  if (n >= N) return;
  float a0 = b2[0], a1 = b2[1];
  const half_t* zr = z + (size_t)n * 256;
  for (int j = 0; j < 256; ++j) {
    float v = (float)zr[j];
    a0 += v * w2[j * 2 + 0];
    a1 += v * w2[j * 2 + 1];
  }
  out[(size_t)n * 2 + 0] = a0;
  out[(size_t)n * 2 + 1] = a1;
}

// ---------------- host-side launch ----------------

extern "C" void kernel_launch(void* const* d_in, const int* in_sizes, int n_in,
                              void* d_out, int out_size, void* d_ws, size_t ws_size,
                              hipStream_t stream) {
  const int N = in_sizes[0] / 256;
  const int E = in_sizes[1] / 2;

  const float* features = (const float*)d_in[0];
  const int* eidx = (const int*)d_in[1];
  const int* erow = eidx;
  const int* ecol = eidx + E;
  const float* enin  = (const float*)d_in[2];
  const float* chebw = (const float*)d_in[3];
  const float* en_w1 = (const float*)d_in[4];
  const float* en_b1 = (const float*)d_in[5];
  const float* en_g1 = (const float*)d_in[6];
  const float* en_be1 = (const float*)d_in[7];
  const float* en_w2 = (const float*)d_in[8];
  const float* en_b2 = (const float*)d_in[9];
  const float* cls_w1 = (const float*)d_in[10];
  const float* cls_b1 = (const float*)d_in[11];
  const float* cls_g = (const float*)d_in[12];
  const float* cls_bb = (const float*)d_in[13];
  const float* cls_w2 = (const float*)d_in[14];
  const float* cls_b2 = (const float*)d_in[15];

  float* logit = (float*)d_out;
  float* ew = logit + (size_t)N * 2;  // edge_weight output slot

  char* ws = (char*)d_ws;
  size_t off = 0;
  auto take = [&](size_t bytes) -> char* {
    char* p = ws + off;
    off = (off + bytes + 255) & ~(size_t)255;
    return p;
  };
  half_t* feat16    = (half_t*)take((size_t)N * 256 * 2);
  half_t* w16t_cheb = (half_t*)take((size_t)4 * 3 * 256 * 256 * 2);  // [L*3][Nout=256][K=256]
  half_t* w16t_en1  = (half_t*)take((size_t)128 * 32 * 2);           // [128][32]
  half_t* w16t_en2  = (half_t*)take((size_t)128 * 128 * 2);          // [128][128]
  half_t* w16t_cls1 = (half_t*)take((size_t)256 * 1024 * 2);         // [256][1024]
  half_t* jk16      = (half_t*)take((size_t)N * 1024 * 2);
  float*  t1_32     = (float*)take((size_t)N * 256 * 4);
  float*  t2_32     = (float*)take((size_t)N * 256 * 4);
  half_t* t1_16     = (half_t*)take((size_t)N * 256 * 2);
  half_t* t2_16     = (half_t*)take((size_t)N * 256 * 2);
  half_t* z16       = (half_t*)take((size_t)N * 256 * 2);
  float*  deg       = (float*)take((size_t)N * 4);
  float*  dis       = (float*)take((size_t)N * 4);
  float*  normv     = (float*)take((size_t)E * 4);

  // feature convert + weight transpose-converts
  {
    long long n = (long long)N * 256;
    k_f32_to_f16<<<(unsigned)((n + 255) / 256), 256, 0, stream>>>(features, feat16, n);
  }
  auto tconv = [&](const float* s, half_t* d, int K, int NC, long long mats) {
    long long total = mats * (long long)K * NC;
    k_transpose_f16<<<(unsigned)((total + 255) / 256), 256, 0, stream>>>(s, d, K, NC, total);
  };
  tconv(chebw, w16t_cheb, 256, 256, 12);  // 12 matrices of 256x256
  tconv(en_w1, w16t_en1, 32, 128, 1);
  tconv(en_w2, w16t_en2, 128, 128, 1);
  tconv(cls_w1, w16t_cls1, 1024, 256, 1);

  // edge encoder -> edge_weight (written directly to its d_out slot)
  int ntiles = (E + 15) / 16;
  k_edge_encoder<<<(ntiles + EW_WAVES - 1) / EW_WAVES, 32 * EW_WAVES, 0, stream>>>(
      enin, w16t_en1, en_b1, en_g1, en_be1, w16t_en2, en_b2, ew, E);

  // graph normalization
  hipMemsetAsync(deg, 0, (size_t)N * 4, stream);
  k_deg_scatter<<<(E + 255) / 256, 256, 0, stream>>>(erow, ew, deg, E);
  k_dis<<<(N + 255) / 256, 256, 0, stream>>>(deg, dis, N);
  k_norm<<<(E + 255) / 256, 256, 0, stream>>>(erow, ecol, ew, dis, normv, E);

  // stacked ChebConv
  const half_t* src16 = feat16;
  int sstride = 256;
  long long propThreads = (long long)E * 64;
  unsigned propBlocks = (unsigned)((propThreads + 255) / 256);
  long long total = (long long)N * 256;
  for (int l = 0; l < 4; ++l) {
    hipMemsetAsync(t1_32, 0, (size_t)N * 256 * 4, stream);
    k_prop_scatter<half_t><<<propBlocks, 256, 0, stream>>>(erow, ecol, normv, src16, sstride, t1_32, E);
    hipMemsetAsync(t2_32, 0, (size_t)N * 256 * 4, stream);
    k_prop_scatter<float><<<propBlocks, 256, 0, stream>>>(erow, ecol, normv, t1_32, 256, t2_32, E);
    k_cheb_finalize<<<(unsigned)((total + 255) / 256), 256, 0, stream>>>(
        t1_32, t2_32, src16, sstride, t1_16, t2_16, total);
    k_cheb_gemm<<<dim3((N + 127) / 128, 16), 256, 0, stream>>>(
        src16, sstride, t1_16, t2_16, w16t_cheb + (size_t)l * 3 * 65536,
        jk16 + (size_t)l * 256, 1024, N);
    src16 = jk16 + (size_t)l * 256;
    sstride = 1024;
  }

  // cls head
  k_cls_gemm<<<dim3((N + 127) / 128, 16), 256, 0, stream>>>(
      jk16, w16t_cls1, cls_b1, cls_g, cls_bb, z16, N);
  k_cls_final<<<(N + 255) / 256, 256, 0, stream>>>(z16, cls_w2, cls_b2, logit, N);
}
